// LightWeightConv1DBlock_53171695125234
// MI455X (gfx1250) — matrix operand order repaired and checked by reading, not compiled, not run on hardware
//
#include <hip/hip_runtime.h>
#include <hip/hip_bf16.h>

// Problem constants
#define Bb   4
#define Tt   1024
#define Cc   1024
#define Hh   16
#define Kk   31
#define Pp   15
#define Rr   64
#define HK   (Hh * Kk)          // 496 = 31 * 16
#define NX   (Bb * Tt * Cc)     // 4194304
#define NW   (HK * Cc)          // 507904
#define NPAD 512                // N padded to 32 tiles of 16 (rows >= 496 are zero)
#define NWP  (NPAD * Cc)        // 524288 padded W elements

typedef __attribute__((ext_vector_type(16))) __bf16 v16bf;
typedef __attribute__((ext_vector_type(8)))  __bf16 v8bf;
typedef __attribute__((ext_vector_type(8)))  float  v8f;

static __device__ __forceinline__ unsigned short f2bf(float f) {
    unsigned int u = __float_as_uint(f);
    u += 0x7FFFu + ((u >> 16) & 1u);   // round-to-nearest-even
    return (unsigned short)(u >> 16);
}

// ---------------------------------------------------------------------------
// Kernel 0: convert x -> bf16, W -> bf16 padded to 512 rows (pad rows zeroed)
// ---------------------------------------------------------------------------
__global__ __launch_bounds__(256) void lwc_convert(const float* __restrict__ x,
                                                   const float* __restrict__ W,
                                                   unsigned short* __restrict__ xbf,
                                                   unsigned short* __restrict__ wbf) {
    long long i = (long long)blockIdx.x * 256 + threadIdx.x;
    if (i < NX) {
        xbf[i] = f2bf(x[i]);
    } else if (i < (long long)NX + NWP) {
        long long j = i - NX;
        wbf[j] = (j < NW) ? f2bf(W[j]) : (unsigned short)0;
    }
}

// ---------------------------------------------------------------------------
// GEMM pipeline stages
//   A fragment: 16x32 bf16 (K 0-7/16-23 in lane-half 0, 8-15/24-31 in half 1)
//   B fragment: 32x16 bf16 (K 0-7/16-23 in lane-half 0, 8-15/24-31 in half 1)
//   pb_wave points at this wave's first N tile; tiles i are +i*8*16 rows,
//   i.e. constant byte offsets 0 / 256K / 512K / 768K (fit 24-bit ioffset).
// ---------------------------------------------------------------------------
static __device__ __forceinline__ void lwc_load_stage(
        const __bf16* __restrict__ pa_base, const __bf16* __restrict__ pb_wave,
        int kc, v8bf& alo, v8bf& ahi, v8bf blo[4], v8bf bhi[4]) {
    const __bf16* pa = pa_base + kc;
    alo = *(const v8bf*)(pa);
    ahi = *(const v8bf*)(pa + 16);
#pragma unroll
    for (int i = 0; i < 4; ++i) {
        const __bf16* pb = pb_wave + (long long)(i * 8 * 16) * Cc + kc;
        blo[i] = *(const v8bf*)(pb);
        bhi[i] = *(const v8bf*)(pb + 16);
    }
}

static __device__ __forceinline__ void lwc_mma_stage(
        v8f acc[4], v8bf alo, v8bf ahi, v8bf blo[4], v8bf bhi[4]) {
    v16bf afrag = __builtin_shufflevector(alo, ahi,
            0, 1, 2, 3, 4, 5, 6, 7, 8, 9, 10, 11, 12, 13, 14, 15);
#pragma unroll
    for (int i = 0; i < 4; ++i) {
        v16bf bfrag = __builtin_shufflevector(blo[i], bhi[i],
                0, 1, 2, 3, 4, 5, 6, 7, 8, 9, 10, 11, 12, 13, 14, 15);
        acc[i] = __builtin_amdgcn_wmma_f32_16x16x32_bf16(
                false, afrag, false, bfrag, (short)0, acc[i], false, false);
    }
}

// ---------------------------------------------------------------------------
// Kernel 1: logits GEMM (bf16 WMMA, f32 accumulate) + softmax over K=31
//   Grid: 256 blocks (one per 16 rows of the 4096-row M space)
//   Block: 256 threads = 8 waves; wave w owns N tiles {w, w+8, w+16, w+24}
//   Two-stage software pipeline: stage n+1 fragments load before stage n's
//   WMMAs issue, so WMMAs don't stall on loadcnt==0.
// ---------------------------------------------------------------------------
__global__ __launch_bounds__(256) void lwc_gemm_softmax(
        const unsigned short* __restrict__ xbf16,
        const unsigned short* __restrict__ wbf16,
        float* __restrict__ wsm) {
    __shared__ float Lg[16 * NPAD];   // 16 x 512 logits, 32 KB

    const int tid  = threadIdx.x;
    const int wave = tid >> 5;
    const int lane = tid & 31;
    const int hl   = lane >> 4;     // lane half: 0 or 1
    const int l16  = lane & 15;

    const int m0 = blockIdx.x * 16;           // row tile in (b*T + t) space
    const int arow = m0 + l16;                // A row this lane carries
    const int abase = hl * 8;                 // K sub-offset for A per lane half

    const __bf16* __restrict__ xb = (const __bf16*)xbf16;
    const __bf16* __restrict__ wb = (const __bf16*)wbf16;

    const __bf16* pa_base = xb + (long long)arow * Cc + abase;
    const __bf16* pb_wave = wb + (long long)(wave * 16 + l16) * Cc + hl * 8;

    v8f acc[4];
#pragma unroll
    for (int i = 0; i < 4; ++i) acc[i] = (v8f){};

    // Ping-pong fragment buffers
    v8bf a0lo, a0hi, b0lo[4], b0hi[4];
    v8bf a1lo, a1hi, b1lo[4], b1hi[4];

    lwc_load_stage(pa_base, pb_wave, 0, a0lo, a0hi, b0lo, b0hi);

#pragma unroll 1
    for (int kc = 0; kc < Cc; kc += 64) {
        lwc_load_stage(pa_base, pb_wave, kc + 32, a1lo, a1hi, b1lo, b1hi);
        lwc_mma_stage(acc, a0lo, a0hi, b0lo, b0hi);
        // Last-iteration prefetch wraps to kc=0: harmless, keeps loop branch-free
        lwc_load_stage(pa_base, pb_wave, (kc + 64) & (Cc - 1), a0lo, a0hi, b0lo, b0hi);
        lwc_mma_stage(acc, a1lo, a1hi, b1lo, b1hi);
    }

    // Scatter D tiles into LDS: D layout M = r + 8*half, N = n0 + l16
#pragma unroll
    for (int i = 0; i < 4; ++i) {
        int n0 = (wave + 8 * i) * 16;
        int mb = hl * 8;
#pragma unroll
        for (int r = 0; r < 8; ++r)
            Lg[(mb + r) * NPAD + n0 + l16] = acc[i][r];
    }
    __syncthreads();

    // Softmax over K=31 per (local row, head): 16*16 = 256 groups, one/thread
    // (only the first 496 columns are read; pad columns ignored)
    const int lt = tid >> 4;
    const int h  = tid & 15;
    const float* row = &Lg[lt * NPAD + h * Kk];

    float mx = -3.402823466e38f;
#pragma unroll
    for (int k = 0; k < Kk; ++k) mx = fmaxf(mx, row[k]);
    float e[Kk];
    float s = 0.f;
#pragma unroll
    for (int k = 0; k < Kk; ++k) { e[k] = __expf(row[k] - mx); s += e[k]; }
    float inv = 1.0f / s;

    float* o = wsm + ((long long)(m0 + lt) * Hh + h) * Kk;
#pragma unroll
    for (int k = 0; k < Kk; ++k) o[k] = e[k] * inv;
}

// ---------------------------------------------------------------------------
// Kernel 2: depthwise band conv with softmaxed dynamic filters (fp32, LDS)
//   Grid: B * (T/64) * H = 1024 blocks; block 256 threads
//   out[b,t,h,r] = sum_k w[b,t,h,k] * x[b, t+k-P, h*64+r]
// ---------------------------------------------------------------------------
__global__ __launch_bounds__(256) void lwc_conv(const float* __restrict__ x,
                                                const float* __restrict__ wsm,
                                                float* __restrict__ out) {
    __shared__ float xs[(64 + Kk - 1) * Rr];  // 94 x 64 fp32 = 23.5 KB
    __shared__ float fs[64 * Kk];             // 64 x 31 fp32 =  7.75 KB

    const int tid = threadIdx.x;
    const int bid = blockIdx.x;
    const int tt = bid & 15;                  // t tile (T/64 = 16)
    const int h  = (bid >> 4) & 15;
    const int b  = bid >> 8;
    const int t0 = tt * 64;

    // Stage x slab: rows t0-15 .. t0+78 (94 rows), cols h*64 .. h*64+63
    for (int i = tid; i < (64 + Kk - 1) * Rr; i += 256) {
        int li = i >> 6;                      // 0..93
        int r  = i & 63;
        int ts = t0 + li - Pp;
        float v = 0.f;
        if (ts >= 0 && ts < Tt)
            v = x[((long long)(b * Tt + ts)) * Cc + h * Rr + r];
        xs[i] = v;
    }
    // Stage filters: 64 x 31
    for (int i = tid; i < 64 * Kk; i += 256) {
        int lt = i / Kk;
        int k  = i - lt * Kk;
        fs[i] = wsm[((long long)(b * Tt + t0 + lt) * Hh + h) * Kk + k];
    }
    __syncthreads();

    const int r   = tid & 63;                 // lanes get consecutive r: no LDS conflicts
    const int ltb = tid >> 6;                 // 0..3, wave-uniform -> filter broadcast
#pragma unroll 4
    for (int j = 0; j < 16; ++j) {
        int lt = ltb * 16 + j;
        float acc = 0.f;
#pragma unroll
        for (int k = 0; k < Kk; ++k)
            acc = fmaf(fs[lt * Kk + k], xs[(lt + k) * Rr + r], acc);
        out[((long long)(b * Tt + t0 + lt)) * Cc + h * Rr + r] = acc;
    }
}

// ---------------------------------------------------------------------------
// Host-side launcher
// ---------------------------------------------------------------------------
extern "C" void kernel_launch(void* const* d_in, const int* in_sizes, int n_in,
                              void* d_out, int out_size, void* d_ws, size_t ws_size,
                              hipStream_t stream) {
    const float* x = (const float*)d_in[0];   // (B,T,C) fp32
    const float* W = (const float*)d_in[1];   // (H*K, C) fp32
    float* out = (float*)d_out;               // (B,T,C) fp32

    // Workspace layout (all offsets 256B aligned):
    //   [0,            8388608)  : xbf  bf16  (NX)
    //   [8388608,      9437184)  : wbf  bf16  (NPAD*Cc, rows >= 496 zero)
    //   [9437184,     17563648)  : wsm  fp32  (B*T*H*K softmaxed filters)
    unsigned short* xbf = (unsigned short*)d_ws;
    unsigned short* wbf = (unsigned short*)((char*)d_ws + (size_t)NX * 2);
    float* wsm = (float*)((char*)d_ws + (size_t)NX * 2 + (size_t)NWP * 2);

    {   // bf16 conversion of x and padded W
        long long n = (long long)NX + NWP;
        int grid = (int)((n + 255) / 256);
        lwc_convert<<<grid, 256, 0, stream>>>(x, W, xbf, wbf);
    }
    {   // WMMA logits GEMM + softmax -> wsm
        lwc_gemm_softmax<<<(Bb * Tt) / 16, 256, 0, stream>>>(xbf, wbf, wsm);
    }
    {   // depthwise dynamic conv
        lwc_conv<<<Bb * Hh * (Tt / 64), 256, 0, stream>>>(x, wsm, out);
    }
}
